// MemoryEfficientTimeEncoding_352187318328
// MI455X (gfx1250) — compile-verified, matrix operand order checked by reading, not run on hardware
//
#include <hip/hip_runtime.h>
#include <hip/hip_bf16.h>

// MI455X (gfx1250) fused time-encoding kernel:
//   tq      = floor(t/0.1)*0.1
//   reduced = relu(cos(tq * w_reduce + b_reduce))      (fp32 math, f16 stored)
//   out     = reduced @ w_expand^T + b_expand          (v_wmma_f32_16x16x32_f16)
//
// B=262144, R=256, D=1024. Output = 1 GiB fp32 -> memory-bound (~46us at 23.3TB/s).
// f16 WMMA keeps the 137 GFLOP GEMM under/at that floor; weights live in L2.

typedef _Float16 v16h __attribute__((ext_vector_type(16)));
typedef _Float16 v8h  __attribute__((ext_vector_type(8)));
typedef _Float16 v4h  __attribute__((ext_vector_type(4)));
typedef float    v8f  __attribute__((ext_vector_type(8)));

#define QUANTF   0.1f
#define R_DIM    256
#define D_DIM    1024
#define M_TILE   256
#define N_TILE   256
#define LDS_K    (R_DIM + 8)        // padded f16 row length (264) -> 528B row, 16B aligned
#define NTHREADS 256

// Load one 16x32 f16 operand fragment (A or B) from LDS in the CDNA5 WMMA
// register layout: lanes 0-15 hold K=0..7 (vgpr0-3) and K=16..23 (vgpr4-7) of
// row (lane&15); lanes 16-31 hold K=8..15 and K=24..31.  p already includes
// the (lane>>4)*8-element half-select; two 16B LDS loads per fragment.
__device__ __forceinline__ v16h load_frag(const _Float16* p) {
    v8h lo = *(const v8h*)(p);
    v8h hi = *(const v8h*)(p + 16);
    return __builtin_shufflevector(lo, hi, 0,1,2,3,4,5,6,7,8,9,10,11,12,13,14,15);
}

__global__ void __launch_bounds__(NTHREADS)
time_encoding_kernel(const float* __restrict__ t,
                     const float* __restrict__ w_reduce_w,
                     const float* __restrict__ w_reduce_b,
                     const float* __restrict__ w_expand_w,
                     const float* __restrict__ w_expand_b,
                     float* __restrict__ out)
{
    extern __shared__ char smem[];
    _Float16* As = (_Float16*)smem;                                   // [M_TILE][LDS_K]
    _Float16* Bs = (_Float16*)(smem + (size_t)M_TILE * LDS_K * 2);    // [N_TILE][LDS_K]
    float*    wr = (float*)(smem + (size_t)(M_TILE + N_TILE) * LDS_K * 2);
    float*    br = wr + R_DIM;

    const int tid = threadIdx.x;
    const int m0g = blockIdx.y * M_TILE;   // batch-row tile
    const int n0g = blockIdx.x * N_TILE;   // output-feature tile

    // ---- stage reduce weights (broadcast-read later, conflict free) ----
    wr[tid] = w_reduce_w[tid];
    br[tid] = w_reduce_b[tid];
    __syncthreads();

    // ---- Phase 1a: A tile, one batch row per thread ----
    {
        float tv = t[m0g + tid];
        float tq = floorf(tv / QUANTF) * QUANTF;
        _Float16* arow = As + tid * LDS_K;
        #pragma unroll 1
        for (int r = 0; r < R_DIM; r += 8) {
            v8h pk;
            #pragma unroll
            for (int j = 0; j < 8; ++j) {
                float c = __cosf(fmaf(tq, wr[r + j], br[r + j]));
                pk[j] = (_Float16)fmaxf(c, 0.0f);
            }
            *(v8h*)(arow + r) = pk;     // ds_store_b128
        }
    }

    // ---- Phase 1b: B tile. GEMM-B(K x N) "column major" == w_expand row-major,
    // so this is a straight coalesced fp32->f16 copy of rows [n0g, n0g+256). ----
    {
        const float4* src = (const float4*)(w_expand_w + (size_t)n0g * R_DIM);
        #pragma unroll 1
        for (int f = tid; f < N_TILE * (R_DIM / 4); f += NTHREADS) {
            float4 v = src[f];                 // consecutive lanes -> consecutive 16B
            int n  = f >> 6;                   // 64 float4 per 256-wide row
            int k4 = (f & 63) << 2;
            v4h pk = { (_Float16)v.x, (_Float16)v.y, (_Float16)v.z, (_Float16)v.w };
            *(v4h*)(Bs + n * LDS_K + k4) = pk; // ds_store_b64
        }
    }
    __syncthreads();

    // ---- Phase 2: 8 waves as 4(M) x 2(N); each wave -> 64 x 128 output ----
    const int lane = tid & 31;
    const int wv   = tid >> 5;
    const int lr   = lane & 15;
    const int hs   = lane >> 4;
    const int wm   = (wv & 3) * 64;     // wave M offset inside tile
    const int wn   = (wv >> 2) * 128;   // wave N offset inside tile

    const _Float16* Ab = As + (wm + lr) * LDS_K + hs * 8;
    const _Float16* Bb = Bs + (wn + lr) * LDS_K + hs * 8;

    v8f acc[4][8] = {};

    #pragma unroll 2
    for (int kc = 0; kc < R_DIM; kc += 32) {
        v16h a[4], b[8];
        #pragma unroll
        for (int mi = 0; mi < 4; ++mi)
            a[mi] = load_frag(Ab + mi * 16 * LDS_K + kc);
        #pragma unroll
        for (int ni = 0; ni < 8; ++ni)
            b[ni] = load_frag(Bb + ni * 16 * LDS_K + kc);
        #pragma unroll
        for (int mi = 0; mi < 4; ++mi)
            #pragma unroll
            for (int ni = 0; ni < 8; ++ni)
                acc[mi][ni] = __builtin_amdgcn_wmma_f32_16x16x32_f16(
                    /*neg_a=*/false, a[mi], /*neg_b=*/false, b[ni],
                    /*c_mod=*/(short)0, acc[mi][ni],
                    /*reuse_a=*/false, /*reuse_b=*/false);
    }

    // ---- Epilogue: bias + store. C layout: vgpr g -> M=g+8*hs, N=lane&15. ----
    float bias[8];
    #pragma unroll
    for (int ni = 0; ni < 8; ++ni)
        bias[ni] = w_expand_b[n0g + wn + ni * 16 + lr];

    #pragma unroll
    for (int mi = 0; mi < 4; ++mi) {
        #pragma unroll
        for (int g = 0; g < 8; ++g) {
            size_t row = (size_t)(m0g + wm + mi * 16 + hs * 8 + g);
            float* orow = out + row * D_DIM + (n0g + wn + lr);
            #pragma unroll
            for (int ni = 0; ni < 8; ++ni)
                orow[ni * 16] = acc[mi][ni][g] + bias[ni];
        }
    }
}

extern "C" void kernel_launch(void* const* d_in, const int* in_sizes, int n_in,
                              void* d_out, int out_size, void* d_ws, size_t ws_size,
                              hipStream_t stream) {
    const float* t    = (const float*)d_in[0];
    const float* wrw  = (const float*)d_in[1];
    const float* wrb  = (const float*)d_in[2];
    const float* wew  = (const float*)d_in[3];
    const float* web  = (const float*)d_in[4];
    float*       out  = (float*)d_out;

    const int B = in_sizes[0];                      // 262144
    const size_t lds_bytes = (size_t)(M_TILE + N_TILE) * LDS_K * 2
                           + 2 * R_DIM * sizeof(float);   // ~266 KB of the 320 KB WGP LDS

    static bool attr_set = false;                   // idempotent, host-side only
    if (!attr_set) {
        hipFuncSetAttribute((const void*)time_encoding_kernel,
                            hipFuncAttributeMaxDynamicSharedMemorySize,
                            (int)lds_bytes);
        attr_set = true;
    }

    dim3 grid(D_DIM / N_TILE, B / M_TILE);          // (4, 1024)
    time_encoding_kernel<<<grid, NTHREADS, lds_bytes, stream>>>(t, wrw, wrb, wew, web, out);
}
